// EnhancedGDN_16965120819901
// MI455X (gfx1250) — compile-verified, matrix operand order checked by reading, not run on hardware
//
#include <hip/hip_runtime.h>
#include <hip/hip_bf16.h>

// ---------------------------------------------------------------------------
// EnhancedGDN forward for MI455X (gfx1250), wave32 + WMMA bf16.
// Temporal attention window == 1  =>  softmax == 1  =>  t_out = data @ v_w^T + v_b (exact).
// Segment softmax folded into unnormalized aggregate + per-node divide.
// ---------------------------------------------------------------------------

typedef __attribute__((ext_vector_type(16))) __bf16 v16bf;
typedef __attribute__((ext_vector_type(8)))  __bf16 v8bf;
typedef __attribute__((ext_vector_type(8)))  float  v8f;

constexpr int   B_   = 64;
constexpr int   N_   = 1000;
constexpr int   D_   = 128;
constexpr int   E_   = 20000;
constexpr int   BN   = B_ * N_;                   // 64000
constexpr long  ETOT = (long)B_ * E_ + BN;        // 1,344,000 edges incl self-loops
constexpr float NEG_SLOPE = 0.2f;
constexpr float BN_EPS    = 1e-5f;

// ---------------- fp32 -> bf16 convert ----------------
__global__ void k_f32_to_bf16(const float* __restrict__ in, __bf16* __restrict__ out, long n) {
  long i = (long)blockIdx.x * blockDim.x + threadIdx.x;
  if (i < n) out[i] = (__bf16)in[i];
}

// ---------------- WMMA GEMM: out[M,Nc] = A[M,K] @ W[Nc,K]^T (+bias) ----------------
// One wave computes a 16x64 strip (4 accumulators) so the A fragment is reused 4x.
// bf16 A/W fragments loaded straight from global in the ISA 16-bit layout:
//   lane<16 : K = k0+[0..7]  and k0+16+[0..7]   (two 16B loads)
//   lane>=16: K = k0+8+[0..7] and k0+24+[0..7]
template<int KDIM, bool OUT_BF16, bool RELU, bool HAS_BIAS>
__global__ __launch_bounds__(256)
void k_wmma_gemm(const __bf16* __restrict__ A, const __bf16* __restrict__ W,
                 const float* __restrict__ bias, void* __restrict__ outp,
                 int M, int Nc, int ldo, int ncol0) {
  constexpr int NT = 4;  // 16-wide tiles per wave
  int wave = (int)(((long)blockIdx.x * blockDim.x + threadIdx.x) >> 5);
  int lane = threadIdx.x & 31;
  int ngroups = Nc / (16 * NT);
  int mtile = wave / ngroups;
  int group = wave - mtile * ngroups;
  if (mtile * 16 >= M) return;            // wave-uniform: EXEC all-1s inside

  int lc  = lane & 15;
  int kh  = (lane >> 4) << 3;             // 0 or 8
  const __bf16* arow = A + (size_t)(mtile * 16 + lc) * KDIM;

  v8f acc[NT];
  #pragma unroll
  for (int t = 0; t < NT; ++t)
    #pragma unroll
    for (int j = 0; j < 8; ++j) acc[t][j] = 0.f;

  #pragma unroll
  for (int k0 = 0; k0 < KDIM; k0 += 32) {
    v16bf a;
    const v8bf a0 = *(const v8bf*)(arow + k0 + kh);
    const v8bf a1 = *(const v8bf*)(arow + k0 + 16 + kh);
    #pragma unroll
    for (int j = 0; j < 8; ++j) { a[j] = a0[j]; a[8 + j] = a1[j]; }
    #pragma unroll
    for (int t = 0; t < NT; ++t) {
      int col = (group * NT + t) * 16 + lc;
      const __bf16* wrow = W + (size_t)col * KDIM;
      v16bf b;
      const v8bf b0 = *(const v8bf*)(wrow + k0 + kh);
      const v8bf b1 = *(const v8bf*)(wrow + k0 + 16 + kh);
      #pragma unroll
      for (int j = 0; j < 8; ++j) { b[j] = b0[j]; b[8 + j] = b1[j]; }
      acc[t] = __builtin_amdgcn_wmma_f32_16x16x32_bf16(
          false, a, false, b, (short)0, acc[t], false, false);
    }
  }

  // C/D layout: lane<16 -> rows r, lane>=16 -> rows 8+r; col = lane&15
  int mbase = mtile * 16 + ((lane >> 4) << 3);
  #pragma unroll
  for (int t = 0; t < NT; ++t) {
    int col = (group * NT + t) * 16 + lc;
    float bv = HAS_BIAS ? bias[col] : 0.f;
    #pragma unroll
    for (int r = 0; r < 8; ++r) {
      float v = acc[t][r] + bv;
      if (RELU) v = fmaxf(v, 0.f);
      size_t o = (size_t)(mbase + r) * ldo + ncol0 + col;
      if (OUT_BF16) ((__bf16*)outp)[o] = (__bf16)v;
      else          ((float*)outp)[o]  = v;
    }
  }
}

// ---------------- per-node attention scores (one wave / node) ----------------
__global__ void k_scores(const float* __restrict__ x, const float* __restrict__ emb,
                         const float* __restrict__ att_i, const float* __restrict__ att_j,
                         const float* __restrict__ att_em_i, const float* __restrict__ att_em_j,
                         float* __restrict__ nsi, float* __restrict__ nsj) {
  int wave = (int)(((long)blockIdx.x * blockDim.x + threadIdx.x) >> 5);
  int lane = threadIdx.x & 31;
  if (wave >= BN) return;
  int n = wave % N_;
  const float4 xv  = *(const float4*)(x   + (size_t)wave * D_ + lane * 4);
  const float4 ev  = *(const float4*)(emb + (size_t)n    * D_ + lane * 4);
  const float4 ai  = *(const float4*)(att_i    + lane * 4);
  const float4 aj  = *(const float4*)(att_j    + lane * 4);
  const float4 aei = *(const float4*)(att_em_i + lane * 4);
  const float4 aej = *(const float4*)(att_em_j + lane * 4);
  float si = xv.x*ai.x + xv.y*ai.y + xv.z*ai.z + xv.w*ai.w
           + ev.x*aei.x + ev.y*aei.y + ev.z*aei.z + ev.w*aei.w;
  float sj = xv.x*aj.x + xv.y*aj.y + xv.z*aj.z + xv.w*aj.w
           + ev.x*aej.x + ev.y*aej.y + ev.z*aej.z + ev.w*aej.w;
  #pragma unroll
  for (int m = 16; m; m >>= 1) { si += __shfl_xor(si, m, 32); sj += __shfl_xor(sj, m, 32); }
  if (lane == 0) { nsi[wave] = si; nsj[wave] = sj; }
}

// monotonic float<->uint mapping for atomicMax on floats
__device__ __forceinline__ unsigned fmap(float f) {
  unsigned u = __float_as_uint(f);
  return (u & 0x80000000u) ? ~u : (u | 0x80000000u);
}
__device__ __forceinline__ float funmap(unsigned u) {
  u = (u & 0x80000000u) ? (u & 0x7FFFFFFFu) : ~u;
  return __uint_as_float(u);
}

__device__ __forceinline__ void edge_decode(long e, const int* __restrict__ ei, int& s, int& d) {
  if (e < (long)B_ * E_) {
    int b = (int)(e / E_);
    int k = (int)(e - (long)b * E_);
    s = ei[k]      + b * N_;   // row0 = src(j)
    d = ei[E_ + k] + b * N_;   // row1 = dst(i)
  } else {
    s = d = (int)(e - (long)B_ * E_);  // self-loops
  }
}

// pass 1: alpha = leakyrelu(nsi[dst] + nsj[src]); segment max over dst
__global__ void k_edge1(const int* __restrict__ ei, const float* __restrict__ nsi,
                        const float* __restrict__ nsj, float* __restrict__ alpha,
                        unsigned* __restrict__ amaxU) {
  long e = (long)blockIdx.x * blockDim.x + threadIdx.x;
  if (e >= ETOT) return;
  int s, d; edge_decode(e, ei, s, d);
  float a = nsi[d] + nsj[s];
  a = (a >= 0.f) ? a : NEG_SLOPE * a;
  alpha[e] = a;
  atomicMax(&amaxU[d], fmap(a));
}

// pass 2: ex = exp(alpha - max); denom[dst]+=ex; agg[dst,:] += ex * x[src,:]
__global__ void k_edge2(const int* __restrict__ ei, const float* __restrict__ alpha,
                        const unsigned* __restrict__ amaxU, const float* __restrict__ x,
                        float* __restrict__ denom, float* __restrict__ agg) {
  long wave = ((long)blockIdx.x * blockDim.x + threadIdx.x) >> 5;
  int lane = threadIdx.x & 31;
  if (wave >= ETOT) return;
  int s, d; edge_decode(wave, ei, s, d);
  float ex = __expf(alpha[wave] - funmap(amaxU[d]));
  if (lane == 0) atomicAdd(&denom[d], ex);
  const float4 xv = *(const float4*)(x + (size_t)s * D_ + lane * 4);
  float* ag = agg + (size_t)d * D_ + lane * 4;
  atomicAdd(ag + 0, ex * xv.x);
  atomicAdd(ag + 1, ex * xv.y);
  atomicAdd(ag + 2, ex * xv.z);
  atomicAdd(ag + 3, ex * xv.w);
}

// normalize agg, add gnn_bias, accumulate per-feature sum / sumsq (block-partial + LDS)
__global__ void k_colstats(float* __restrict__ agg, const float* __restrict__ denom,
                           const float* __restrict__ gbias,
                           float* __restrict__ csum, float* __restrict__ csumsq) {
  __shared__ float s_s[256], s_q[256];
  int d    = threadIdx.x & 127;
  int half = threadIdx.x >> 7;
  int rend = (blockIdx.x + 1) * 128;
  float gs = 0.f, gq = 0.f;
  float gb = gbias[d];
  for (int r = blockIdx.x * 128 + half; r < rend; r += 2) {
    size_t off = (size_t)r * D_ + d;
    float v = agg[off] / denom[r] + gb;
    agg[off] = v;
    gs += v; gq += v * v;
  }
  s_s[threadIdx.x] = gs; s_q[threadIdx.x] = gq;
  __syncthreads();
  if (half == 0) {
    atomicAdd(&csum[d],   s_s[d] + s_s[d + 128]);
    atomicAdd(&csumsq[d], s_q[d] + s_q[d + 128]);
  }
}

// batchnorm (batch stats, biased var) + relu -> bf16 into combined[:,0:D]
__global__ void k_bn(const float* __restrict__ agg, const float* __restrict__ csum,
                     const float* __restrict__ csumsq, const float* __restrict__ gamma,
                     const float* __restrict__ beta, __bf16* __restrict__ combined) {
  long idx = (long)blockIdx.x * blockDim.x + threadIdx.x;
  if (idx >= (long)BN * D_) return;
  int  d   = (int)(idx & (D_ - 1));
  long row = idx >> 7;
  float mu  = csum[d]   * (1.0f / BN);
  float var = csumsq[d] * (1.0f / BN) - mu * mu;
  float v = gamma[d] * (agg[idx] - mu) * rsqrtf(var + BN_EPS) + beta[d];
  combined[row * (2 * D_) + d] = (__bf16)fmaxf(v, 0.f);
}

// out[i] = dot(fused[i,:], out_w) + out_b   (one wave / node)
__global__ void k_out(const float* __restrict__ fused, const float* __restrict__ ow,
                      const float* __restrict__ ob, float* __restrict__ out) {
  int wave = (int)(((long)blockIdx.x * blockDim.x + threadIdx.x) >> 5);
  int lane = threadIdx.x & 31;
  if (wave >= BN) return;
  const float4 f = *(const float4*)(fused + (size_t)wave * D_ + lane * 4);
  const float4 w = *(const float4*)(ow + lane * 4);
  float s = f.x*w.x + f.y*w.y + f.z*w.z + f.w*w.w;
  #pragma unroll
  for (int m = 16; m; m >>= 1) s += __shfl_xor(s, m, 32);
  if (lane == 0) out[wave] = s + ob[0];
}

// ---------------------------------------------------------------------------
extern "C" void kernel_launch(void* const* d_in, const int* in_sizes, int n_in,
                              void* d_out, int out_size, void* d_ws, size_t ws_size,
                              hipStream_t stream) {
  (void)in_sizes; (void)n_in; (void)out_size; (void)ws_size;
  const float* data      = (const float*)d_in[0];
  const int*   ei        = (const int*)  d_in[1];
  const float* emb       = (const float*)d_in[2];
  const float* lin_w     = (const float*)d_in[3];
  const float* att_i     = (const float*)d_in[4];
  const float* att_j     = (const float*)d_in[5];
  const float* att_em_i  = (const float*)d_in[6];
  const float* att_em_j  = (const float*)d_in[7];
  const float* gnn_bias  = (const float*)d_in[8];
  const float* bn_gamma  = (const float*)d_in[9];
  const float* bn_beta   = (const float*)d_in[10];
  // d_in[11..14] = q_w,q_b,k_w,k_b and d_in[17]=temperature: exactly cancelled (softmax of singleton)
  const float* v_w       = (const float*)d_in[15];
  const float* v_b       = (const float*)d_in[16];
  const float* f_w1      = (const float*)d_in[18];
  const float* f_b1      = (const float*)d_in[19];
  const float* f_w2      = (const float*)d_in[20];
  const float* f_b2      = (const float*)d_in[21];
  const float* out_w     = (const float*)d_in[22];
  const float* out_b     = (const float*)d_in[23];
  float* out = (float*)d_out;

  // ---- workspace carve-up (256B aligned) ----
  char* base = (char*)d_ws;
  size_t off = 0;
  auto alloc = [&](size_t bytes) -> void* {
    off = (off + 255) & ~(size_t)255;
    void* r = base + off;
    off += bytes;
    return r;
  };
  __bf16* dataB = (__bf16*)alloc((size_t)BN * D_ * 2);       // reused later as hidden
  __bf16* hB    = dataB;
  __bf16* combB = (__bf16*)alloc((size_t)BN * 2 * D_ * 2);   // [s_out | t_out] bf16
  __bf16* linB  = (__bf16*)alloc((size_t)D_ * D_ * 2);
  __bf16* vB    = (__bf16*)alloc((size_t)D_ * D_ * 2);
  __bf16* f1B   = (__bf16*)alloc((size_t)D_ * 2 * D_ * 2);
  __bf16* f2B   = (__bf16*)alloc((size_t)D_ * D_ * 2);
  float*  x     = (float*)alloc((size_t)BN * D_ * 4);        // reused later as fused
  float*  fused = x;
  size_t zbeg = (off + 255) & ~(size_t)255;                  // contiguous zeroed region:
  float*    agg    = (float*)   alloc((size_t)BN * D_ * 4);
  float*    denom  = (float*)   alloc((size_t)BN * 4);
  unsigned* amaxU  = (unsigned*)alloc((size_t)BN * 4);
  float*    csum   = (float*)   alloc((size_t)D_ * 4);
  float*    csumsq = (float*)   alloc((size_t)D_ * 4);
  size_t zend = off;
  float* alpha = (float*)alloc((size_t)ETOT * 4);
  float* nsi   = (float*)alloc((size_t)BN * 4);
  float* nsj   = (float*)alloc((size_t)BN * 4);

  hipMemsetAsync(base + zbeg, 0, zend - zbeg, stream);

  auto cgrid = [](long n) { return (int)((n + 255) / 256); };
  k_f32_to_bf16<<<cgrid((long)BN * D_), 256, 0, stream>>>(data,  dataB, (long)BN * D_);
  k_f32_to_bf16<<<cgrid(D_ * D_),     256, 0, stream>>>(lin_w, linB, D_ * D_);
  k_f32_to_bf16<<<cgrid(D_ * D_),     256, 0, stream>>>(v_w,   vB,   D_ * D_);
  k_f32_to_bf16<<<cgrid(D_ * 2 * D_), 256, 0, stream>>>(f_w1,  f1B,  D_ * 2 * D_);
  k_f32_to_bf16<<<cgrid(D_ * D_),     256, 0, stream>>>(f_w2,  f2B,  D_ * D_);

  // waves = (M/16) * (Nc/64); 8 waves per 256-thread block
  const int gemmBlocks = ((BN / 16) * (D_ / 64) * 32 + 255) / 256;  // 1000

  // x = data @ lin_w^T (f32, no bias)
  k_wmma_gemm<128, false, false, false><<<gemmBlocks, 256, 0, stream>>>(
      dataB, linB, nullptr, x, BN, D_, D_, 0);
  // t_out = data @ v_w^T + v_b  -> combined[:, D:2D] (bf16)
  k_wmma_gemm<128, true, false, true><<<gemmBlocks, 256, 0, stream>>>(
      dataB, vB, v_b, combB, BN, D_, 2 * D_, D_);

  k_scores<<<(BN * 32 + 255) / 256, 256, 0, stream>>>(
      x, emb, att_i, att_j, att_em_i, att_em_j, nsi, nsj);
  k_edge1<<<(int)((ETOT + 255) / 256), 256, 0, stream>>>(ei, nsi, nsj, alpha, amaxU);
  k_edge2<<<(int)((ETOT * 32 + 255) / 256), 256, 0, stream>>>(ei, alpha, amaxU, x, denom, agg);
  k_colstats<<<BN / 128, 256, 0, stream>>>(agg, denom, gnn_bias, csum, csumsq);
  k_bn<<<(int)(((long)BN * D_ + 255) / 256), 256, 0, stream>>>(
      agg, csum, csumsq, bn_gamma, bn_beta, combB);

  // hidden = relu(combined @ f_w1^T + f_b1)  (bf16, K=256)  [overwrites dataB]
  k_wmma_gemm<256, true, true, true><<<gemmBlocks, 256, 0, stream>>>(
      combB, f1B, f_b1, hB, BN, D_, D_, 0);
  // fused = hidden @ f_w2^T + f_b2 (f32)  [overwrites x]
  k_wmma_gemm<128, false, false, true><<<gemmBlocks, 256, 0, stream>>>(
      hB, f2B, f_b2, fused, BN, D_, D_, 0);

  k_out<<<(BN * 32 + 255) / 256, 256, 0, stream>>>(fused, out_w, out_b, out);
}
